// Upformer_35940286333138
// MI455X (gfx1250) — matrix operand order repaired and checked by reading, not compiled, not run on hardware
//
#include <hip/hip_runtime.h>

typedef __attribute__((ext_vector_type(16))) _Float16 v16h;
typedef __attribute__((ext_vector_type(8)))  _Float16 v8h;
typedef __attribute__((ext_vector_type(8)))  float    v8f;
typedef __attribute__((ext_vector_type(4)))  int      v4i;

#define NPIX   131072      // 8*128*128
#define RSQRT8 0.35355339059327373f

// workspace layout (float indices)
#define WS_WFUSET  0       // _Float16[96][64] fused weights, fragment-order (12288 B = 3072 floats)
#define WS_WOT     3072    // _Float16[32][64] Wo transposed, fragment-order (4096 B = 1024 floats)
#define WS_BLOGIN  6144    // f32[32] logit bias, in-bounds window positions
#define WS_BLOGOUT 6176    // f32[32] logit bias, zero-padded positions
#define WS_BVVIN   6208    // f32[64] v bias, in-bounds
#define WS_BVVOUT  6272    // f32[64] v bias, padded
#define WS_PV      16384   // f32[NPIX][96] projected features (48MB, L2-resident: 192MB L2)

// ---------------------------------------------------------------------------
// Kernel 1: fold Wk@Wka·q, Wv@Wva and all biases into fused weights; emit the
// WMMA B-operands pre-converted to f16 in fragment (column-major) order.
// ---------------------------------------------------------------------------
__global__ __launch_bounds__(256) void prep_kernel(
    const float* __restrict__ uq, const float* __restrict__ Wk, const float* __restrict__ bk,
    const float* __restrict__ Wv, const float* __restrict__ bv,
    const float* __restrict__ Wq, const float* __restrict__ bq,
    const float* __restrict__ Wka, const float* __restrict__ bka,
    const float* __restrict__ Wva, const float* __restrict__ bva,
    const float* __restrict__ Wo, float* __restrict__ ws)
{
    __shared__ float wkk[64 * 64];
    __shared__ float wvv[64 * 64];
    __shared__ float wlog[64 * 32];
    __shared__ float qvec[4 * 64];
    __shared__ float kbi[64], kbo[64];
    const int t = threadIdx.x;

    // Wkk = Wk@Wka, Wvv = Wv@Wva (both to LDS)
    for (int idx = t; idx < 4096; idx += 256) {
        int c = idx >> 6, j = idx & 63;
        float sk = 0.f, sv = 0.f;
        for (int e = 0; e < 64; ++e) {
            sk += Wk[c * 64 + e] * Wka[e * 64 + j];
            sv += Wv[c * 64 + e] * Wva[e * 64 + j];
        }
        wkk[idx] = sk;
        wvv[idx] = sv;
    }
    // qvec = up_query @ Wq + bq  [4][64]
    if (t < 256) {
        int q = t >> 6, j = t & 63;
        float s = bq[j];
        for (int c = 0; c < 64; ++c) s += uq[q * 64 + c] * Wq[c * 64 + j];
        qvec[t] = s;
    }
    // K-path biases: in-bounds gets bk twice (to_k + aux-zeros branch), padded gets once
    if (t < 64) {
        float s = 0.f;
        for (int d = 0; d < 64; ++d) s += bk[d] * Wka[d * 64 + t];
        kbi[t] = 2.f * s + bka[t];
        kbo[t] = s + bka[t];
        float sv = bva[t];
        for (int d = 0; d < 64; ++d) sv += bv[d] * Wva[d * 64 + t];
        ws[WS_BVVIN + t] = sv;
        ws[WS_BVVOUT + t] = bva[t];
    }
    __syncthreads();
    // Wlogit[c][h*4+q] = (1/sqrt(8)) * sum_d Wkk[c][h*8+d] * qvec[q][h*8+d]
    for (int idx = t; idx < 2048; idx += 256) {
        int c = idx >> 5, j2 = idx & 31;
        int h = j2 >> 2, q = j2 & 3;
        float s = 0.f;
        for (int d = 0; d < 8; ++d) s += wkk[c * 64 + h * 8 + d] * qvec[q * 64 + h * 8 + d];
        wlog[c * 32 + j2] = s * RSQRT8;
    }
    if (t < 32) {
        int h = t >> 2, q = t & 3;
        float si = 0.f, so = 0.f;
        for (int d = 0; d < 8; ++d) {
            si += kbi[h * 8 + d] * qvec[q * 64 + h * 8 + d];
            so += kbo[h * 8 + d] * qvec[q * 64 + h * 8 + d];
        }
        ws[WS_BLOGIN + t]  = si * RSQRT8;
        ws[WS_BLOGOUT + t] = so * RSQRT8;
    }
    __syncthreads();
    // WfuseT[n][k]: column-major f16 so a B fragment is 16 contiguous halves
    _Float16* wfT = (_Float16*)(ws + WS_WFUSET);
    for (int idx = t; idx < 6144; idx += 256) {
        int n = idx >> 6, k = idx & 63;
        float val = (n < 32) ? wlog[k * 32 + n] : wvv[k * 64 + (n - 32)];
        wfT[idx] = (_Float16)val;
    }
    // WoT[n][k] f16 for the out-projection B fragments
    _Float16* woT = (_Float16*)(ws + WS_WOT);
    for (int idx = t; idx < 2048; idx += 256) {
        int n = idx >> 6, k = idx & 63;
        woT[idx] = (_Float16)Wo[k * 32 + n];
    }
}

// ---------------------------------------------------------------------------
// Kernel 2: PV[N,96] = x[N,64] @ Wfuse[64,96]. WfuseT (12KB f16) is staged to
// LDS once per workgroup (async global->LDS DMA, ASYNCcnt-tracked) and shared
// by all 8 waves; each wave does a 16-pixel x 96-col tile = 12 WMMAs.
// ---------------------------------------------------------------------------
__global__ __launch_bounds__(256) void proj_kernel(
    const float* __restrict__ x, const float* __restrict__ ws, float* __restrict__ pv)
{
    __shared__ _Float16 wfs[96 * 64];   // 12KB, 32B-aligned fragments
    {
        const int t = threadIdx.x;
#if __has_builtin(__builtin_amdgcn_global_load_async_to_lds_b128)
        // CDNA5 async DMA path: GLOBAL_LOAD_ASYNC_TO_LDS_B128
        // signature (probe-confirmed arity 4): (v4i AS1* src, v4i AS3* dst, imm off, imm cpol)
        typedef __attribute__((address_space(1))) v4i* gas_t;
        typedef __attribute__((address_space(3))) v4i* las_t;
        char* src = (char*)(ws + WS_WFUSET);
        char* dst = (char*)wfs;
        for (int i = t; i < 768; i += 256) {
            v4i* sp = (v4i*)(src + i * 16);
            v4i* dp = (v4i*)(dst + i * 16);
            __builtin_amdgcn_global_load_async_to_lds_b128((gas_t)sp, (las_t)dp, 0, 0);
        }
#if __has_builtin(__builtin_amdgcn_s_wait_asynccnt)
        __builtin_amdgcn_s_wait_asynccnt(0);
#else
        asm volatile("s_wait_asynccnt 0" ::: "memory");
#endif
#else
        const uint4* src = (const uint4*)(ws + WS_WFUSET);
        uint4* dst = (uint4*)wfs;
        for (int i = t; i < 768; i += 256) dst[i] = src[i];
#endif
    }
    __syncthreads();

    const int tid  = blockIdx.x * 256 + threadIdx.x;
    const int wave = tid >> 5;
    const int lane = tid & 31;
    const int hi   = lane >> 4;          // half-wave select (ISA 7.12.2 A/B layout)
    const int mb   = wave * 16;
    const int m    = mb + (lane & 15);   // pixel (GEMM row) for A fragment
    // x is [B,64,128,128]: channel stride 16384
    const float* xb = x + (size_t)(m >> 14) * 64 * 16384 + (m & 16383);

    // A fragment 16x32 f16: lanes<16 hold K 0..15, lanes>=16 K 16..31
    v16h a0, a1;
#pragma unroll
    for (int h = 0; h < 16; ++h) {
        int k = hi * 16 + h;
        a0[h] = (_Float16)xb[(size_t)k * 16384];
        a1[h] = (_Float16)xb[(size_t)(k + 32) * 16384];
    }
#pragma unroll
    for (int nt = 0; nt < 6; ++nt) {
        int n = nt * 16 + (lane & 15);
        // B fragment: 16 contiguous halves per lane from LDS (2x ds_read_b128)
        v16h b0 = *(const v16h*)(wfs + n * 64 + hi * 16);
        v16h b1 = *(const v16h*)(wfs + n * 64 + 32 + hi * 16);
        v8f c = {};
        c = __builtin_amdgcn_wmma_f32_16x16x32_f16(false, a0, false, b0, (short)0, c, false, false);
        c = __builtin_amdgcn_wmma_f32_16x16x32_f16(false, a1, false, b1, (short)0, c, false, false);
#pragma unroll
        for (int r = 0; r < 8; ++r) {
            int row = mb + r + hi * 8;               // C/D: M = r + 8*(lane>=16)
            pv[(size_t)row * 96 + nt * 16 + (lane & 15)] = c[r];
        }
    }
}

// ---------------------------------------------------------------------------
// Kernel 3: per-pixel 3x3 attention (32 lanes <-> 32 (head,query) rows),
// WMMA out-projection (16 rows = 4px x 4 queries), LayerNorm, pixel-shuffle.
// ---------------------------------------------------------------------------
__global__ __launch_bounds__(256) void attn_kernel(
    const float* __restrict__ ws, const float* __restrict__ pv,
    const float* __restrict__ rel_pos, const float* __restrict__ bo,
    const float* __restrict__ ln_g, const float* __restrict__ ln_b,
    float* __restrict__ out)
{
    __shared__ _Float16 outl[8][16 * 64];  // attention output rows (f16: WMMA A input)
    __shared__ float dproj[8][16 * 32];    // projected rows per wave
    const int wl   = threadIdx.x >> 5;
    const int lane = threadIdx.x & 31;
    const int pixBase = (blockIdx.x * 8 + wl) * 4;
    const int h = lane >> 2, q = lane & 3;  // lane <-> (head, query)

    float rel[9];
#pragma unroll
    for (int p = 0; p < 9; ++p) rel[p] = rel_pos[h * 36 + q * 9 + p];
    const float bli = ws[WS_BLOGIN + lane];
    const float blo = ws[WS_BLOGOUT + lane];
    float bvi[8], bvo[8];
#pragma unroll
    for (int d = 0; d < 8; ++d) {
        bvi[d] = ws[WS_BVVIN + h * 8 + d];
        bvo[d] = ws[WS_BVVOUT + h * 8 + d];
    }

    for (int px = 0; px < 4; ++px) {
        int n = pixBase + px;
        int b = n >> 14, hw = n & 16383;
        int y = hw >> 7, xw = hw & 127;
        float logit[9];
        int nbr[9];
#pragma unroll
        for (int p = 0; p < 9; ++p) {
            int dy = p / 3 - 1, dx = p % 3 - 1;
            int yn = y + dy, xn = xw + dx;
            bool inb = ((unsigned)yn < 128u) && ((unsigned)xn < 128u);
            int nn = (b << 14) + (yn << 7) + xn;
            nbr[p] = inb ? nn : -1;
            logit[p] = inb ? (pv[(size_t)nn * 96 + lane] + bli + rel[p]) : (blo + rel[p]);
        }
        float mx = logit[0];
#pragma unroll
        for (int p = 1; p < 9; ++p) mx = fmaxf(mx, logit[p]);
        float e[9], sum = 0.f;
#pragma unroll
        for (int p = 0; p < 9; ++p) { e[p] = __expf(logit[p] - mx); sum += e[p]; }
        float inv = 1.f / sum;
        float acc[8] = {0.f, 0.f, 0.f, 0.f, 0.f, 0.f, 0.f, 0.f};
#pragma unroll
        for (int p = 0; p < 9; ++p) {
            float a = e[p] * inv;
            if (nbr[p] >= 0) {
                float vbuf[8];
                const float4* vr = (const float4*)(pv + (size_t)nbr[p] * 96 + 32 + h * 8);
                *(float4*)&vbuf[0] = vr[0];
                *(float4*)&vbuf[4] = vr[1];
#pragma unroll
                for (int d = 0; d < 8; ++d) acc[d] += a * (vbuf[d] + bvi[d]);
            } else {
#pragma unroll
                for (int d = 0; d < 8; ++d) acc[d] += a * bvo[d];
            }
        }
        int row = px * 4 + q;
        v8h pk;
#pragma unroll
        for (int d = 0; d < 8; ++d) pk[d] = (_Float16)acc[d];
        *(v8h*)(&outl[wl][row * 64 + h * 8]) = pk;   // one ds_store_b128
    }
    __syncthreads();

    // out projection: D[16,32] = A(outl 16x64) @ B(WoT 64x32) + bo, via 4 WMMAs
    const int hi = lane >> 4;
    const int ml = lane & 15;
    const _Float16* op = outl[wl];
    v8h c00 = *(const v8h*)(op + ml * 64 + hi * 16);
    v8h c01 = *(const v8h*)(op + ml * 64 + 8 + hi * 16);
    v8h c10 = *(const v8h*)(op + ml * 64 + 32 + hi * 16);
    v8h c11 = *(const v8h*)(op + ml * 64 + 40 + hi * 16);
    v16h a0, a1;
#pragma unroll
    for (int i = 0; i < 8; ++i) {
        a0[i] = c00[i]; a0[i + 8] = c01[i];
        a1[i] = c10[i]; a1[i + 8] = c11[i];
    }
    const _Float16* woT = (const _Float16*)(ws + WS_WOT);
#pragma unroll
    for (int nt = 0; nt < 2; ++nt) {
        int nn = nt * 16 + ml;
        v16h b0 = *(const v16h*)(woT + nn * 64 + hi * 16);
        v16h b1 = *(const v16h*)(woT + nn * 64 + 32 + hi * 16);
        float bias = bo[nn];
        v8f c;
#pragma unroll
        for (int r = 0; r < 8; ++r) c[r] = bias;     // bias preloaded into accumulator
        c = __builtin_amdgcn_wmma_f32_16x16x32_f16(false, a0, false, b0, (short)0, c, false, false);
        c = __builtin_amdgcn_wmma_f32_16x16x32_f16(false, a1, false, b1, (short)0, c, false, false);
#pragma unroll
        for (int r = 0; r < 8; ++r) dproj[wl][(r + hi * 8) * 32 + nn] = c[r];
    }
    __syncthreads();

    // LayerNorm over 32 + pixel-shuffle store: lanes 0..15 each own one row
    if (lane < 16) {
        int row = lane;
        int n = pixBase + (row >> 2);
        int qq = row & 3;
        int b = n >> 14, hw = n & 16383;
        int y = hw >> 7, xw = hw & 127;
        int sy = qq >> 1, sx = qq & 1;
        float v[32];
        float mu = 0.f;
#pragma unroll
        for (int o = 0; o < 32; ++o) { v[o] = dproj[wl][row * 32 + o]; mu += v[o]; }
        mu *= (1.f / 32.f);
        float var = 0.f;
#pragma unroll
        for (int o = 0; o < 32; ++o) { float d = v[o] - mu; var += d * d; }
        var *= (1.f / 32.f);
        float rstd = rsqrtf(var + 1e-5f);
        size_t base = (size_t)b * 32 * 65536 + (size_t)(y * 2 + sy) * 256 + (xw * 2 + sx);
#pragma unroll
        for (int o = 0; o < 32; ++o) {
            out[base + (size_t)o * 65536] = (v[o] - mu) * rstd * ln_g[o] + ln_b[o];
        }
    }
}

// ---------------------------------------------------------------------------
extern "C" void kernel_launch(void* const* d_in, const int* in_sizes, int n_in,
                              void* d_out, int out_size, void* d_ws, size_t ws_size,
                              hipStream_t stream) {
    const float* x       = (const float*)d_in[0];
    const float* uq      = (const float*)d_in[1];
    const float* Wk      = (const float*)d_in[2];
    const float* bk      = (const float*)d_in[3];
    const float* Wv      = (const float*)d_in[4];
    const float* bv      = (const float*)d_in[5];
    const float* Wq      = (const float*)d_in[6];
    const float* bq      = (const float*)d_in[7];
    const float* Wka     = (const float*)d_in[8];
    const float* bka     = (const float*)d_in[9];
    const float* Wva     = (const float*)d_in[10];
    const float* bva     = (const float*)d_in[11];
    const float* Wo      = (const float*)d_in[12];
    const float* bo      = (const float*)d_in[13];
    const float* rel_pos = (const float*)d_in[14];
    const float* ln_g    = (const float*)d_in[15];
    const float* ln_b    = (const float*)d_in[16];
    float* out = (float*)d_out;
    float* ws  = (float*)d_ws;          // needs ~48.1 MB (PV intermediate)
    float* pv  = ws + WS_PV;

    prep_kernel<<<1, 256, 0, stream>>>(uq, Wk, bk, Wv, bv, Wq, bq, Wka, bka, Wva, bva, Wo, ws);
    // NPIX/16 = 8192 waves -> 1024 blocks of 8 waves
    proj_kernel<<<1024, 256, 0, stream>>>(x, ws, pv);
    // NPIX/4 = 32768 waves -> 4096 blocks of 8 waves
    attn_kernel<<<4096, 256, 0, stream>>>(ws, pv, rel_pos, bo, ln_g, ln_b, out);
}
